// MultiHeadSelfAttention_13451837571696
// MI455X (gfx1250) — compile-verified, matrix operand order checked by reading, not run on hardware
//
#include <hip/hip_runtime.h>

typedef __attribute__((ext_vector_type(16))) _Float16 v16h;
typedef __attribute__((ext_vector_type(8)))  _Float16 v8h;
typedef __attribute__((ext_vector_type(8)))  float    v8f;

#define D_MODEL  1024
#define N_HEADS  16
#define HEAD_DIM 64
#define BATCH    2
#define SEQ      2048
#define MROWS    (BATCH * SEQ)   // 4096
#define KCHUNK   64

union Frag { v16h v; v8h h[2]; };

static __device__ __forceinline__ v8f wmma_f16(const Frag& a, const Frag& b, v8f c) {
  // D = A(16x32 f16) * B(32x16 f16) + C(16x16 f32)
  return __builtin_amdgcn_wmma_f32_16x16x32_f16(false, a.v, false, b.v, (short)0, c, false, false);
}

// ---------------------------------------------------------------------------
// f32 -> f16 elementwise convert
// ---------------------------------------------------------------------------
__global__ void __launch_bounds__(256)
convert_f32_f16(const float* __restrict__ in, _Float16* __restrict__ out, int n) {
  for (int i = blockIdx.x * blockDim.x + threadIdx.x; i < n; i += gridDim.x * blockDim.x)
    out[i] = (_Float16)in[i];
}

// ---------------------------------------------------------------------------
// Convert + transpose 1024x1024 weight:  out[n][k] = (f16) in[k][n]
// ---------------------------------------------------------------------------
__global__ void __launch_bounds__(256)
convert_transpose(const float* __restrict__ in, _Float16* __restrict__ out) {
  int n = blockIdx.x * 16 + threadIdx.x;
  int k = blockIdx.y * 16 + threadIdx.y;
  out[(size_t)n * D_MODEL + k] = (_Float16)in[(size_t)k * D_MODEL + n];
}

// ---------------------------------------------------------------------------
// LDS-tiled WMMA GEMM:  C = A[M][K](f16) * Bt[N][K](f16 pre-transposed) + bias
// Block = 8 waves = 128(M) x 64(N) tile. Bt slab staged in LDS, K chunks of 64,
// double-buffered, one barrier per chunk. Each wave: 16(M) x 64(N), 4 accs.
// mode 0: f16 row-major [M][N]     mode 1: f16 Vt[((b*16+h)*64+d)][s]
// mode 2: f32 row-major [M][N]
// ---------------------------------------------------------------------------
__global__ void __launch_bounds__(256)
wmma_gemm(const _Float16* __restrict__ A, const _Float16* __restrict__ Bt,
          const float* __restrict__ bias, void* __restrict__ Cout,
          int Mdim, int Ndim, int Kdim, int mode)
{
  __shared__ _Float16 lb[2][64][KCHUNK];     // 2 x 8KB B tile [col][k]

  const int tid   = threadIdx.x;
  const int lane  = tid & 31;
  const int wave  = tid >> 5;
  const int tilesN = Ndim >> 6;
  const int bm    = blockIdx.x / tilesN;     // 128-row block tile
  const int bn    = blockIdx.x % tilesN;     // 64-col block tile

  const int l15   = lane & 15;
  const int hi    = lane >> 4;
  const int khalf = hi * 8;

  // Cooperative B staging: 64 rows x 128B = 8KB; each thread owns 32B.
  const int brow  = tid >> 2;                // 0..63
  const int bkoff = (tid & 3) << 4;          // 0,16,32,48 (halves)
  const _Float16* bsrc = Bt + (size_t)(bn * 64 + brow) * Kdim + bkoff;

  // This wave's A rows
  const _Float16* arow = A + (size_t)(bm * 128 + wave * 16 + l15) * Kdim + khalf;

  v8f acc0 = {}, acc1 = {}, acc2 = {}, acc3 = {};

  // Preload chunk 0
  {
    v8h r0 = *(const v8h*)(bsrc);
    v8h r1 = *(const v8h*)(bsrc + 8);
    *(v8h*)(&lb[0][brow][bkoff])     = r0;
    *(v8h*)(&lb[0][brow][bkoff + 8]) = r1;
  }
  __syncthreads();

  const int nk = Kdim / KCHUNK;
  for (int c = 0; c < nk; ++c) {
    const int buf = c & 1;
    v8h r0, r1;
    if (c + 1 < nk) {                        // issue next-chunk global loads
      r0 = *(const v8h*)(bsrc + (c + 1) * KCHUNK);
      r1 = *(const v8h*)(bsrc + (c + 1) * KCHUNK + 8);
    }

    #pragma unroll
    for (int s = 0; s < 2; ++s) {            // two K-steps of 32
      const int k0 = c * KCHUNK + s * 32;
      Frag a;
      a.h[0] = *(const v8h*)(arow + k0);
      a.h[1] = *(const v8h*)(arow + k0 + 16);
      const int kl = s * 32 + khalf;
      Frag b0, b1, b2, b3;
      b0.h[0] = *(const v8h*)(&lb[buf][     l15][kl]);
      b0.h[1] = *(const v8h*)(&lb[buf][     l15][kl + 16]);
      b1.h[0] = *(const v8h*)(&lb[buf][16 + l15][kl]);
      b1.h[1] = *(const v8h*)(&lb[buf][16 + l15][kl + 16]);
      b2.h[0] = *(const v8h*)(&lb[buf][32 + l15][kl]);
      b2.h[1] = *(const v8h*)(&lb[buf][32 + l15][kl + 16]);
      b3.h[0] = *(const v8h*)(&lb[buf][48 + l15][kl]);
      b3.h[1] = *(const v8h*)(&lb[buf][48 + l15][kl + 16]);
      acc0 = wmma_f16(a, b0, acc0);
      acc1 = wmma_f16(a, b1, acc1);
      acc2 = wmma_f16(a, b2, acc2);
      acc3 = wmma_f16(a, b3, acc3);
    }

    if (c + 1 < nk) {                        // stage next chunk (after compute)
      *(v8h*)(&lb[buf ^ 1][brow][bkoff])     = r0;
      *(v8h*)(&lb[buf ^ 1][brow][bkoff + 8]) = r1;
    }
    __syncthreads();
  }

  #pragma unroll
  for (int j = 0; j < 4; ++j) {
    v8f acc = (j == 0) ? acc0 : (j == 1) ? acc1 : (j == 2) ? acc2 : acc3;
    const int n  = bn * 64 + j * 16 + l15;
    const float bv = bias ? bias[n] : 0.0f;
    #pragma unroll
    for (int r = 0; r < 8; ++r) {
      const int m = bm * 128 + wave * 16 + r + hi * 8;
      const float val = acc[r] + bv;
      if (mode == 2) {
        ((float*)Cout)[(size_t)m * Ndim + n] = val;
      } else if (mode == 0) {
        ((_Float16*)Cout)[(size_t)m * Ndim + n] = (_Float16)val;
      } else {
        const int bb = m >> 11, s = m & 2047;
        const int hh = n >> 6,  d = n & 63;
        ((_Float16*)Cout)[(((size_t)(bb * N_HEADS + hh) * HEAD_DIM + d) << 11) + s]
            = (_Float16)val;
      }
    }
  }
}

// ---------------------------------------------------------------------------
// Flash-style causal attention, block-shared K/V through LDS.
// Block = 8 waves = 8 consecutive 16-query tiles of one (b,h).
// K/V 32-key chunks double-buffered in LDS, shared by all 8 waves.
// Q,K: f16 [4096][1024]; Vt: f16 [B*H*64][2048]; O: f16 [4096][1024]
// ---------------------------------------------------------------------------
__global__ void __launch_bounds__(256)
attn_kernel(const _Float16* __restrict__ Q, const _Float16* __restrict__ K,
            const _Float16* __restrict__ Vt, _Float16* __restrict__ O)
{
  __shared__ _Float16 lk[2][32][HEAD_DIM];  // [buf][key][d]   2 x 4KB
  __shared__ _Float16 lv[2][HEAD_DIM][32];  // [buf][d][key]   2 x 4KB
  __shared__ float    sc[8][16][32];        // per-wave scores 16KB
  __shared__ _Float16 pf[8][16][32];        // per-wave exp(P)  8KB
  __shared__ float    rsc[8][16];
  __shared__ float    rsm[8][16];

  const int tid  = threadIdx.x;
  const int lane = tid & 31;
  const int wave = tid >> 5;
  const int grp  = blockIdx.x & 15;         // q-tile group (8 tiles)
  const int h    = (blockIdx.x >> 4) & 15;
  const int b    = blockIdx.x >> 8;
  const int qbase = grp * 8;
  const int qt   = qbase + wave;            // this wave's q-tile

  const int l15   = lane & 15;
  const int hi    = lane >> 4;
  const int khalf = hi * 8;

  // Q fragments: d = 0..31 and 32..63
  const _Float16* qrow = Q + (size_t)((b << 11) + qt * 16 + l15) * D_MODEL + h * HEAD_DIM;
  Frag qa0, qa1;
  qa0.h[0] = *(const v8h*)(qrow + khalf);
  qa0.h[1] = *(const v8h*)(qrow + khalf + 16);
  qa1.h[0] = *(const v8h*)(qrow + 32 + khalf);
  qa1.h[1] = *(const v8h*)(qrow + 32 + khalf + 16);

  // Cooperative K staging: 32 keys x 128B; thread owns one 16B chunk.
  const int krow = tid >> 3;                // 0..31 (key within chunk)
  const int kofs = (tid & 7) << 3;          // 0..56 (halves within head_dim)
  const _Float16* ksrc = K + (size_t)((b << 11) + krow) * D_MODEL + h * HEAD_DIM + kofs;
  // Cooperative V staging: 64 d-rows x 64B; thread owns one 16B chunk.
  const int vrow = tid >> 2;                // 0..63 (d)
  const int vofs = (tid & 3) << 3;          // 0..24 (keys within chunk)
  const _Float16* vsrc = Vt + ((size_t)(b * N_HEADS + h) * HEAD_DIM + vrow) * SEQ + vofs;

  v8f o0 = {}, o1 = {}, o2 = {}, o3 = {};
  float row_m = -1e30f, row_s = 0.0f;       // lanes 0..15 own row = lane
  const float scale = 0.125f;               // 1/sqrt(64)
  const int my_nch  = (qt * 16 + 47) >> 5;              // chunks this wave needs
  const int nch_blk = ((qbase + 7) * 16 + 47) >> 5;     // block-uniform max

  // Preload chunk 0
  {
    v8h rk = *(const v8h*)(ksrc);
    v8h rv = *(const v8h*)(vsrc);
    *(v8h*)(&lk[0][krow][kofs]) = rk;
    *(v8h*)(&lv[0][vrow][vofs]) = rv;
  }
  __syncthreads();

  for (int c = 0; c < nch_blk; ++c) {
    const int buf = c & 1;
    v8h rk, rv;
    if (c + 1 < nch_blk) {                  // issue next-chunk global loads
      rk = *(const v8h*)(ksrc + (size_t)(c + 1) * 32 * D_MODEL);
      rv = *(const v8h*)(vsrc + (c + 1) * 32);
    }

    if (c < my_nch) {                       // wave-uniform: EXEC full inside
      const int key0 = c * 32;

      // ---- scores: 16q x 32k via 4 WMMAs, masked+scaled into LDS ----
      #pragma unroll
      for (int t = 0; t < 2; ++t) {
        const int lkey = t * 16 + l15;
        Frag kb0, kb1;
        kb0.h[0] = *(const v8h*)(&lk[buf][lkey][khalf]);
        kb0.h[1] = *(const v8h*)(&lk[buf][lkey][khalf + 16]);
        kb1.h[0] = *(const v8h*)(&lk[buf][lkey][32 + khalf]);
        kb1.h[1] = *(const v8h*)(&lk[buf][lkey][32 + khalf + 16]);
        v8f s = {};
        s = wmma_f16(qa0, kb0, s);
        s = wmma_f16(qa1, kb1, s);
        #pragma unroll
        for (int r = 0; r < 8; ++r) {
          const int m = r + hi * 8;
          const int q = qt * 16 + m;
          sc[wave][m][t * 16 + l15] = (key0 + lkey <= q) ? s[r] * scale : -1e30f;
        }
      }
      asm volatile("s_wait_dscnt 0" ::: "memory");

      // ---- online softmax (lanes 0..15, one row each) ----
      if (lane < 16) {
        const int row = lane;
        float cm = -1e30f;
        #pragma unroll
        for (int j = 0; j < 32; ++j) cm = fmaxf(cm, sc[wave][row][j]);
        const float nm = fmaxf(row_m, cm);
        const float f  = __expf(row_m - nm);
        float ps = 0.0f;
        #pragma unroll
        for (int j = 0; j < 32; ++j) {
          const float p = __expf(sc[wave][row][j] - nm);
          ps += p;
          pf[wave][row][j] = (_Float16)p;
        }
        row_s = row_s * f + ps;
        row_m = nm;
        rsc[wave][row] = f;
      }
      asm volatile("s_wait_dscnt 0" ::: "memory");

      // ---- rescale O accumulators ----
      #pragma unroll
      for (int r = 0; r < 8; ++r) {
        const float f = rsc[wave][r + hi * 8];
        o0[r] *= f; o1[r] *= f; o2[r] *= f; o3[r] *= f;
      }

      // ---- P A-fragment from LDS (cross-lane transpose) ----
      Frag pa;
      pa.h[0] = *(const v8h*)(&pf[wave][l15][khalf]);
      pa.h[1] = *(const v8h*)(&pf[wave][l15][khalf + 16]);

      // ---- O += P * V  (4 WMMAs over d sub-tiles) ----
      #pragma unroll
      for (int j = 0; j < 4; ++j) {
        Frag vb;
        vb.h[0] = *(const v8h*)(&lv[buf][j * 16 + l15][khalf]);
        vb.h[1] = *(const v8h*)(&lv[buf][j * 16 + l15][khalf + 16]);
        if      (j == 0) o0 = wmma_f16(pa, vb, o0);
        else if (j == 1) o1 = wmma_f16(pa, vb, o1);
        else if (j == 2) o2 = wmma_f16(pa, vb, o2);
        else             o3 = wmma_f16(pa, vb, o3);
      }
    }

    if (c + 1 < nch_blk) {                  // stage next chunk after compute
      *(v8h*)(&lk[buf ^ 1][krow][kofs]) = rk;
      *(v8h*)(&lv[buf ^ 1][vrow][vofs]) = rv;
    }
    __syncthreads();
  }

  // ---- normalize and store ----
  if (lane < 16) rsm[wave][lane] = row_s;
  asm volatile("s_wait_dscnt 0" ::: "memory");

  #pragma unroll
  for (int r = 0; r < 8; ++r) {
    const int m = r + hi * 8;
    const float inv = 1.0f / rsm[wave][m];
    _Float16* op = O + (size_t)((b << 11) + qt * 16 + m) * D_MODEL + h * HEAD_DIM;
    op[      l15] = (_Float16)(o0[r] * inv);
    op[16 +  l15] = (_Float16)(o1[r] * inv);
    op[32 +  l15] = (_Float16)(o2[r] * inv);
    op[48 +  l15] = (_Float16)(o3[r] * inv);
  }
}

// ---------------------------------------------------------------------------
extern "C" void kernel_launch(void* const* d_in, const int* in_sizes, int n_in,
                              void* d_out, int out_size, void* d_ws, size_t ws_size,
                              hipStream_t stream)
{
  const float* x  = (const float*)d_in[0];
  const float* Wq = (const float*)d_in[1];
  const float* bq = (const float*)d_in[2];
  const float* Wk = (const float*)d_in[3];
  const float* bk = (const float*)d_in[4];
  const float* Wv = (const float*)d_in[5];
  const float* bv = (const float*)d_in[6];
  const float* Wo = (const float*)d_in[7];
  const float* bo = (const float*)d_in[8];

  char* ws = (char*)d_ws;
  _Float16* xh  = (_Float16*)(ws);                      // 8 MB  x f16
  _Float16* wqT = (_Float16*)(ws + ((size_t) 8 << 20)); // 2 MB
  _Float16* wkT = (_Float16*)(ws + ((size_t)10 << 20)); // 2 MB
  _Float16* wvT = (_Float16*)(ws + ((size_t)12 << 20)); // 2 MB
  _Float16* woT = (_Float16*)(ws + ((size_t)14 << 20)); // 2 MB
  _Float16* qh  = (_Float16*)(ws + ((size_t)16 << 20)); // 8 MB
  _Float16* kh  = (_Float16*)(ws + ((size_t)24 << 20)); // 8 MB
  _Float16* vt  = (_Float16*)(ws + ((size_t)32 << 20)); // 8 MB  V transposed/head
  _Float16* oh  = (_Float16*)(ws + ((size_t)40 << 20)); // 8 MB  attention out

  convert_f32_f16<<<1024, 256, 0, stream>>>(x, xh, MROWS * D_MODEL);

  dim3 tb(16, 16), tg(64, 64);
  convert_transpose<<<tg, tb, 0, stream>>>(Wq, wqT);
  convert_transpose<<<tg, tb, 0, stream>>>(Wk, wkT);
  convert_transpose<<<tg, tb, 0, stream>>>(Wv, wvT);
  convert_transpose<<<tg, tb, 0, stream>>>(Wo, woT);

  const int blocksG = (MROWS / 128) * (D_MODEL / 64);   // 512
  wmma_gemm<<<blocksG, 256, 0, stream>>>(xh, wqT, bq, qh, MROWS, D_MODEL, D_MODEL, 0);
  wmma_gemm<<<blocksG, 256, 0, stream>>>(xh, wkT, bk, kh, MROWS, D_MODEL, D_MODEL, 0);
  wmma_gemm<<<blocksG, 256, 0, stream>>>(xh, wvT, bv, vt, MROWS, D_MODEL, D_MODEL, 1);

  attn_kernel<<<512, 256, 0, stream>>>(qh, kh, vt, oh);

  wmma_gemm<<<blocksG, 256, 0, stream>>>(oh, woT, bo, (float*)d_out,
                                         MROWS, D_MODEL, D_MODEL, 2);
}
